// DeepseekV3MoECalibrate_10084583211681
// MI455X (gfx1250) — matrix operand order repaired and checked by reading, not compiled, not run on hardware
//
#include <hip/hip_runtime.h>
#include <hip/hip_bf16.h>
#include <math.h>

// ---------------------------------------------------------------------------
// DeepseekV3 MoE (calibrate-all-experts) for gfx1250 / MI455X
//   T=1024 tokens, H=1024, I=512 per routed expert (E=32), shared I2=1024
//   Heavy GEMMs -> v_wmma_f32_16x16x32_bf16 (wave32 WMMA), double-buffered
//   LDS, A-tiles staged by the Tensor Data Mover (tensor_load_to_lds).
// ---------------------------------------------------------------------------

#define T_TOK 1024
#define HDIM  1024
#define IDIM  512
#define I2DIM 1024
#define NEXP  32

typedef __attribute__((ext_vector_type(16))) __bf16 bf16x16;
typedef __attribute__((ext_vector_type(8)))  __bf16 bf16x8;
typedef __attribute__((ext_vector_type(8)))  float  floatx8;
typedef __attribute__((ext_vector_type(4)))  unsigned int uint32x4;
typedef __attribute__((ext_vector_type(8)))  int          int32x8;
typedef __attribute__((ext_vector_type(4)))  int          int32x4;

#define BM 128
#define BN 64
#define KT 64
#define LPAD 8
#define LSTR (KT + LPAD)   // LDS row stride in halves (144B: bank-spread, 16B aligned)

static __device__ inline bf16x16 cat8(bf16x8 lo, bf16x8 hi) {
  bf16x16 r;
#pragma unroll
  for (int j = 0; j < 8; ++j) { r[j] = lo[j]; r[8 + j] = hi[j]; }
  return r;
}

static __device__ inline floatx8 wmma_bf16(bf16x16 a, bf16x16 b, floatx8 c) {
  return __builtin_amdgcn_wmma_f32_16x16x32_bf16(false, a, false, b,
                                                 (short)0, c, false, false);
}

// TDM: 2D tile load (bf16 elements) global -> LDS with hardware row padding:
// pad_interval=4 -> pad after every 32 DWORDs (=128B = one 64-half row),
// pad_amount=3   -> insert 4 DWORDs (=16B = LPAD halves). LDS row stride 144B.
static __device__ inline void tdm_load_tile_bf16(unsigned ldsAddr,
                                                 const __bf16* gsrc,
                                                 int tileK, int tileM,
                                                 int rowStrideElems) {
  unsigned long long ga = (unsigned long long)(uintptr_t)gsrc;
  uint32x4 g0;
  g0[0] = 1u;                                        // count=1 (valid user D#)
  g0[1] = ldsAddr;                                   // lds_addr [63:32]
  g0[2] = (unsigned)(ga & 0xffffffffu);              // global_addr lo
  g0[3] = (unsigned)((ga >> 32) & 0x01ffffffu) | (2u << 30);  // addr[56:32]|type=2
  int32x8 g1;
  g1[0] = (1 << 16) | (1 << 20) | (4 << 22) | (3 << 25);  // data_size=2B,pad_en,int=32dw,amt=4dw
  g1[1] = (tileK & 0xffff) << 16;                    // tensor_dim0[15:0]
  g1[2] = (tileM & 0xffff) << 16;                    // tensor_dim0[31:16]=0 | tensor_dim1[15:0]
  g1[3] = (tileK & 0xffff) << 16;                    // tensor_dim1[31:16]=0 | tile_dim0
  g1[4] = (tileM & 0xffff);                          // tile_dim1 | tile_dim2=0
  g1[5] = rowStrideElems;                            // tensor_dim0_stride[31:0]
  g1[6] = 0;                                         // stride hi | dim1_stride lo
  g1[7] = 0;
  int32x4 z4 = {0, 0, 0, 0};
  int32x8 z8 = {0, 0, 0, 0, 0, 0, 0, 0};
  __builtin_amdgcn_tensor_load_to_lds(g0, g1, z4, z4, z8, 0);
}

// --------------------------- fp32 -> bf16 convert ---------------------------
__global__ void cvt_kernel(const float* __restrict__ src, __bf16* __restrict__ dst, int n) {
  int i = (blockIdx.x * 256 + threadIdx.x) * 4;
  if (i + 3 < n) {
    float4 v = *(const float4*)(src + i);
    dst[i + 0] = (__bf16)v.x; dst[i + 1] = (__bf16)v.y;
    dst[i + 2] = (__bf16)v.z; dst[i + 3] = (__bf16)v.w;
  }
}

// --------------------------- router / gating --------------------------------
__global__ void gate_kernel(const float* __restrict__ x,
                            const float* __restrict__ gw,
                            const float* __restrict__ gb,
                            float* __restrict__ wdense) {
  int lane = threadIdx.x & 31;
  int wid  = threadIdx.x >> 5;
  int t    = blockIdx.x * 8 + wid;

  __shared__ float s_s[8][32];
  __shared__ float s_c[8][32];
  __shared__ float s_w[8][32];

  const float* xr = x  + (size_t)t    * HDIM;
  const float* wr = gw + (size_t)lane * HDIM;
  float acc = 0.f;
#pragma unroll 4
  for (int h = 0; h < HDIM; h += 4) {
    float4 xv = *(const float4*)(xr + h);
    float4 wv = *(const float4*)(wr + h);
    acc += xv.x * wv.x + xv.y * wv.y + xv.z * wv.z + xv.w * wv.w;
  }
  float s  = 1.f / (1.f + expf(-acc));
  s_s[wid][lane] = s;
  s_c[wid][lane] = s + gb[lane];
  s_w[wid][lane] = 0.f;
  __syncthreads();

  if (lane == 0) {
    float gs[4];
#pragma unroll
    for (int g = 0; g < 4; ++g) {
      float m1 = -1e30f, m2 = -1e30f;
      for (int j = 0; j < 8; ++j) {
        float v = s_c[wid][g * 8 + j];
        if (v > m1) { m2 = m1; m1 = v; } else if (v > m2) { m2 = v; }
      }
      gs[g] = m1 + m2;
    }
    int g1 = 0;
    for (int g = 1; g < 4; ++g) if (gs[g] > gs[g1]) g1 = g;
    int g2 = -1;
    for (int g = 0; g < 4; ++g) if (g != g1 && (g2 < 0 || gs[g] > gs[g2])) g2 = g;

    int  chosen[8];
    bool used[32];
    for (int e = 0; e < 32; ++e) used[e] = false;
    for (int k = 0; k < 8; ++k) {
      int best = 0; float bv = -1e30f;
      for (int e = 0; e < 32; ++e) {
        int g = e >> 3;
        if (!used[e] && (g == g1 || g == g2)) {
          float v = s_c[wid][e];
          if (v > bv) { bv = v; best = e; }
        }
      }
      used[best] = true;
      chosen[k]  = best;
    }
    float wsum = 0.f;
    for (int k = 0; k < 8; ++k) wsum += s_s[wid][chosen[k]];
    float inv = 2.5f / (wsum + 1e-20f);
    for (int k = 0; k < 8; ++k) s_w[wid][chosen[k]] = s_s[wid][chosen[k]] * inv;
  }
  __syncthreads();
  wdense[(size_t)t * NEXP + lane] = s_w[wid][lane];
}

// --------------------------- gate+up fused GEMM -----------------------------
// act[e][t][i] = bf16( silu(x@Wg_e^T) * (x@Wu_e^T) )
// Double-buffered LDS; A tile staged by TDM (wave 0), B tiles fp32->bf16 by
// all waves. Block 256 thr = 8 waves (4 M x 2 N), block tile 128x64, KT=64.
__global__ void gateup_kernel(const __bf16* __restrict__ xbf,
                              const float* __restrict__ gateW,
                              const float* __restrict__ upW,
                              __bf16* __restrict__ actBase,
                              int nI, size_t wStride, size_t aStride) {
  __shared__ __bf16 As[2][BM * LSTR];
  __shared__ __bf16 Bg[2][BN * LSTR];
  __shared__ __bf16 Bu[2][BN * LSTR];

  const int e = blockIdx.y;
  const float* wg = gateW + (size_t)e * wStride;
  const float* wu = upW   + (size_t)e * wStride;
  __bf16* act     = actBase + (size_t)e * aStride;

  const int ntN  = nI / BN;
  const int row0 = (blockIdx.x / ntN) * BM;
  const int col0 = (blockIdx.x % ntN) * BN;

  const int tid  = threadIdx.x;
  const int lane = tid & 31;
  const int wid  = tid >> 5;
  const int wm   = wid & 3;
  const int wn   = wid >> 2;

  floatx8 accG[2][2] = {{{0.f}}};
  floatx8 accU[2][2] = {{{0.f}}};

  auto stageA = [&](int buf, int k0) {  // TDM: one wave issues whole tile
    if (wid == 0)
      tdm_load_tile_bf16((unsigned)(uintptr_t)(&As[buf][0]),
                         xbf + (size_t)row0 * HDIM + k0, KT, BM, HDIM);
  };
  auto stageB = [&](int buf, int k0) {  // fp32 -> bf16 convert while staging
    for (int c = tid; c < BN * (KT / 4); c += 256) {
      int r  = c / (KT / 4);
      int kc = (c % (KT / 4)) * 4;
      size_t off = (size_t)(col0 + r) * HDIM + k0 + kc;
      float4 vg = *(const float4*)(wg + off);
      float4 vu = *(const float4*)(wu + off);
      __bf16* pg = &Bg[buf][r * LSTR + kc];
      __bf16* pu = &Bu[buf][r * LSTR + kc];
      pg[0] = (__bf16)vg.x; pg[1] = (__bf16)vg.y; pg[2] = (__bf16)vg.z; pg[3] = (__bf16)vg.w;
      pu[0] = (__bf16)vu.x; pu[1] = (__bf16)vu.y; pu[2] = (__bf16)vu.z; pu[3] = (__bf16)vu.w;
    }
  };

  // prologue: stage 0 into buffer 0
  stageA(0, 0);
  stageB(0, 0);
  if (wid == 0) __builtin_amdgcn_s_wait_tensorcnt(0);
  __syncthreads();

  const int mlo = lane & 15, khalf = (lane >> 4) * 8;
  const int kb  = (lane >> 4) * 16;

  for (int k0 = 0; k0 < HDIM; k0 += KT) {
    const int buf = (k0 / KT) & 1;
    // prefetch next stage into the alternate buffer while computing
    if (k0 + KT < HDIM) {
      stageA(buf ^ 1, k0 + KT);
      stageB(buf ^ 1, k0 + KT);
    }
#pragma unroll
    for (int ks = 0; ks < KT; ks += 32) {
      bf16x16 afr[2];
#pragma unroll
      for (int mt = 0; mt < 2; ++mt) {
        const __bf16* pa = &As[buf][(wm * 32 + mt * 16 + mlo) * LSTR + ks];
        afr[mt] = cat8(*(const bf16x8*)(pa + kb),
                       *(const bf16x8*)(pa + kb + 8));
      }
#pragma unroll
      for (int nt = 0; nt < 2; ++nt) {
        int n = wn * 32 + nt * 16 + mlo;
        bf16x16 bg = *(const bf16x16*)(&Bg[buf][n * LSTR + ks + kb]);
        bf16x16 bu = *(const bf16x16*)(&Bu[buf][n * LSTR + ks + kb]);
#pragma unroll
        for (int mt = 0; mt < 2; ++mt) {
          accG[mt][nt] = wmma_bf16(afr[mt], bg, accG[mt][nt]);
          accU[mt][nt] = wmma_bf16(afr[mt], bu, accU[mt][nt]);
        }
      }
    }
    if (wid == 0) __builtin_amdgcn_s_wait_tensorcnt(0);
    __syncthreads();
  }

  // epilogue: silu(g)*u -> bf16 act
  const int nlo = lane & 15;
  const int hi8 = (lane >> 4) * 8;
#pragma unroll
  for (int mt = 0; mt < 2; ++mt)
#pragma unroll
    for (int nt = 0; nt < 2; ++nt)
#pragma unroll
      for (int r = 0; r < 8; ++r) {
        float g = accG[mt][nt][r];
        float u = accU[mt][nt][r];
        float h = (g / (1.f + expf(-g))) * u;
        int t = row0 + wm * 32 + mt * 16 + hi8 + r;
        int i = col0 + wn * 32 + nt * 16 + nlo;
        act[(size_t)t * nI + i] = (__bf16)h;
      }
}

// --------------------------- down GEMM (+fused combine) ---------------------
// out[t,h] (+)= sum_e  (w[t,e] * act_e[t,:]) @ down_e[h,:]^T
// Combine weight folded into A staging -> one accumulator chain, no atomics.
// Double-buffered over flattened (expert, k) stages.
template <bool ACCUM>
__global__ void down_kernel(const __bf16* __restrict__ actBase,
                            const float* __restrict__ downBase,
                            const float* __restrict__ scale,   // [T,NEXP] or null
                            int nExp, int Kdim,
                            size_t aStride, size_t wStride,
                            float* __restrict__ out) {
  __shared__ __bf16 As[2][BM * LSTR];
  __shared__ __bf16 Bs[2][BN * LSTR];

  const int ntN  = HDIM / BN;
  const int row0 = (blockIdx.x / ntN) * BM;
  const int col0 = (blockIdx.x % ntN) * BN;

  const int tid  = threadIdx.x;
  const int lane = tid & 31;
  const int wid  = tid >> 5;
  const int wm   = wid & 3;
  const int wn   = wid >> 2;

  const int kStages = Kdim / KT;
  const int nStages = nExp * kStages;

  floatx8 acc[2][2] = {{{0.f}}};

  auto stage = [&](int s, int buf) {
    int e  = s / kStages;
    int k0 = (s % kStages) * KT;
    const __bf16* actp = actBase + (size_t)e * aStride;
    const float*  wd   = downBase + (size_t)e * wStride;
    for (int c = tid; c < BM * (KT / 8); c += 256) {
      int r  = c / (KT / 8);
      int kc = (c % (KT / 8)) * 8;
      bf16x8 v = *(const bf16x8*)(actp + (size_t)(row0 + r) * Kdim + k0 + kc);
      if (scale) {
        float sw = scale[(size_t)(row0 + r) * NEXP + e];
#pragma unroll
        for (int j = 0; j < 8; ++j) v[j] = (__bf16)((float)v[j] * sw);
      }
      *(bf16x8*)(&As[buf][r * LSTR + kc]) = v;
    }
    for (int c = tid; c < BN * (KT / 4); c += 256) {
      int r  = c / (KT / 4);
      int kc = (c % (KT / 4)) * 4;
      float4 v = *(const float4*)(wd + (size_t)(col0 + r) * Kdim + k0 + kc);
      __bf16* pb = &Bs[buf][r * LSTR + kc];
      pb[0] = (__bf16)v.x; pb[1] = (__bf16)v.y; pb[2] = (__bf16)v.z; pb[3] = (__bf16)v.w;
    }
  };

  stage(0, 0);
  __syncthreads();

  const int mlo = lane & 15, khalf = (lane >> 4) * 8;
  const int kb  = (lane >> 4) * 16;

  for (int s = 0; s < nStages; ++s) {
    const int buf = s & 1;
    if (s + 1 < nStages) stage(s + 1, buf ^ 1);
#pragma unroll
    for (int ks = 0; ks < KT; ks += 32) {
      bf16x16 afr[2];
#pragma unroll
      for (int mt = 0; mt < 2; ++mt) {
        const __bf16* pa = &As[buf][(wm * 32 + mt * 16 + mlo) * LSTR + ks];
        afr[mt] = cat8(*(const bf16x8*)(pa + kb),
                       *(const bf16x8*)(pa + kb + 8));
      }
#pragma unroll
      for (int nt = 0; nt < 2; ++nt) {
        bf16x16 bfr = *(const bf16x16*)(&Bs[buf][(wn * 32 + nt * 16 + mlo) * LSTR + ks + kb]);
#pragma unroll
        for (int mt = 0; mt < 2; ++mt)
          acc[mt][nt] = wmma_bf16(afr[mt], bfr, acc[mt][nt]);
      }
    }
    __syncthreads();
  }

  const int nlo = lane & 15;
  const int hi8 = (lane >> 4) * 8;
#pragma unroll
  for (int mt = 0; mt < 2; ++mt)
#pragma unroll
    for (int nt = 0; nt < 2; ++nt)
#pragma unroll
      for (int r = 0; r < 8; ++r) {
        int t = row0 + wm * 32 + mt * 16 + hi8 + r;
        int h = col0 + wn * 32 + nt * 16 + nlo;
        size_t idx = (size_t)t * HDIM + h;
        float v = acc[mt][nt][r];
        if (ACCUM) out[idx] = out[idx] + v;
        else       out[idx] = v;
      }
}

// ---------------------------------------------------------------------------
extern "C" void kernel_launch(void* const* d_in, const int* in_sizes, int n_in,
                              void* d_out, int out_size, void* d_ws, size_t ws_size,
                              hipStream_t stream) {
  const float* x  = (const float*)d_in[0];   // [2,512,1024]
  const float* gw = (const float*)d_in[1];   // [32,1024]
  const float* gb = (const float*)d_in[2];   // [32]
  const float* eg = (const float*)d_in[3];   // [32,512,1024]
  const float* eu = (const float*)d_in[4];   // [32,512,1024]
  const float* ed = (const float*)d_in[5];   // [32,1024,512]
  const float* sg = (const float*)d_in[6];   // [1024,1024]
  const float* su = (const float*)d_in[7];   // [1024,1024]
  const float* sd = (const float*)d_in[8];   // [1024,1024]
  float* out = (float*)d_out;                // [2,512,1024]

  char* ws = (char*)d_ws;
  __bf16* xbf    = (__bf16*)(ws);                                   // 2 MB
  __bf16* act    = (__bf16*)(ws + ((size_t)2  << 20));              // 32 MB
  __bf16* actSh  = (__bf16*)(ws + ((size_t)34 << 20));              // 2 MB
  float*  wdense = (float*) (ws + ((size_t)36 << 20));              // 128 KB

  // 1) x -> bf16
  cvt_kernel<<<(T_TOK * HDIM) / (256 * 4), 256, 0, stream>>>(x, xbf, T_TOK * HDIM);

  // 2) router -> dense combine weights [T,32]
  gate_kernel<<<T_TOK / 8, 256, 0, stream>>>(x, gw, gb, wdense);

  // 3) routed experts gate+up
  gateup_kernel<<<dim3((T_TOK / BM) * (IDIM / BN), NEXP), 256, 0, stream>>>(
      xbf, eg, eu, act, IDIM, (size_t)IDIM * HDIM, (size_t)T_TOK * IDIM);

  // 4) shared expert gate+up (I2 = 1024)
  gateup_kernel<<<dim3((T_TOK / BM) * (I2DIM / BN), 1), 256, 0, stream>>>(
      xbf, sg, su, actSh, I2DIM, 0, 0);

  // 5) shared down -> store into out
  down_kernel<false><<<(T_TOK / BM) * (HDIM / BN), 256, 0, stream>>>(
      actSh, sd, nullptr, 1, I2DIM, 0, 0, out);

  // 6) routed down with fused top-k combine -> accumulate into out
  down_kernel<true><<<(T_TOK / BM) * (HDIM / BN), 256, 0, stream>>>(
      act, ed, wdense, NEXP, IDIM, (size_t)T_TOK * IDIM, (size_t)HDIM * IDIM, out);
}
